// ScaledDotProductAttention_58841051955917
// MI455X (gfx1250) — matrix operand order repaired and checked by reading, not compiled, not run on hardware
//
#include <hip/hip_runtime.h>

#define NB 32
#define LQ 2048
#define LK 2048
#define DD 128

typedef __attribute__((ext_vector_type(16))) __bf16 v16bf;
typedef __attribute__((ext_vector_type(8)))  __bf16 v8bf;
typedef __attribute__((ext_vector_type(4)))  __bf16 v4bf;
typedef __attribute__((ext_vector_type(2)))  __bf16 v2bf;
typedef __attribute__((ext_vector_type(8)))  float  v8f;

static __device__ __forceinline__ __bf16 f2bf(float f) {
    // round-to-nearest-even fp32 -> bf16 (fallback path)
    unsigned u = __float_as_uint(f);
    unsigned r = ((u >> 16) & 1u) + 0x7fffu;
    unsigned short h = (unsigned short)((u + r) >> 16);
    union { unsigned short s; __bf16 b; } cv; cv.s = h;
    return cv.b;
}

#if __has_builtin(__builtin_amdgcn_cvt_pk_bf16_f32)
static __device__ __forceinline__ v2bf pk2bf(float a, float b) {
    return __builtin_amdgcn_cvt_pk_bf16_f32(a, b);   // v_cvt_pk_bf16_f32
}
#else
static __device__ __forceinline__ v2bf pk2bf(float a, float b) {
    v2bf r; r[0] = f2bf(a); r[1] = f2bf(b); return r;
}
#endif

#if __has_builtin(__builtin_amdgcn_exp2f)
static __device__ __forceinline__ float fast_exp2(float x) {
    return __builtin_amdgcn_exp2f(x);                // raw v_exp_f32
}
#else
static __device__ __forceinline__ float fast_exp2(float x) { return exp2f(x); }
#endif

__global__ __launch_bounds__(256)
__attribute__((amdgpu_waves_per_eu(4)))        // <=256 VGPRs: no msb churn
void attn_wmma_kernel(const float* __restrict__ qp,
                      const float* __restrict__ kp,
                      const float* __restrict__ vp,
                      const unsigned char* __restrict__ maskp,
                      float* __restrict__ outp)
{
    __shared__ __bf16 kbuf[32 * 128];        // [key][d] row-major, 8 KB
    __shared__ __bf16 vbuf[128 * 32];        // [d][key] transposed, 8 KB
    __shared__ __bf16 pbuf[8][32 * 32];      // per-wave P staging, 16 KB
    __shared__ __bf16 qlds[8 * 8 * 32 * 16]; // per-wave Q fragments, 64 KB

    const int tid  = threadIdx.x;
    const int wave = tid >> 5;
    const int lane = tid & 31;
    const int l16  = lane & 15;
    const int hi   = lane >> 4;             // 0: lanes 0-15, 1: lanes 16-31

    const int qblocks = LQ / 256;           // 8 blocks of 256 q-rows per batch
    const int b     = blockIdx.x / qblocks;
    const int qb    = blockIdx.x % qblocks;
    const int qbase = qb * 256 + wave * 32; // this wave's 32 q rows

    // scale' = log2(e)/sqrt(128): S is pre-multiplied by log2e so the
    // softmax exp(x) becomes a bare v_exp_f32 (exp2), no per-element mul.
    const float scale = 1.4426950408889634f * 0.08838834764831845f;

    // ---- Load Q tile (32x128), convert to bf16 A fragments, park in LDS ----
    // A layout (16-bit, 16x32): M = lane%16; per lane two contiguous 8-elem
    // regions: [d0 + hi*8 .. +7] and [d0+16 + hi*8 .. +7].
    // qlds slot: fragment f = t*4+c, 32 bytes per lane.
#pragma unroll
    for (int t = 0; t < 2; ++t) {
        const float* qrow = qp + ((size_t)b * LQ + qbase + t * 16 + l16) * DD;
        const int K0 = hi * 8;
#pragma unroll
        for (int c = 0; c < 4; ++c) {
            const int d0 = c * 32;
            float4 f0 = *(const float4*)(qrow + d0 + K0);
            float4 f1 = *(const float4*)(qrow + d0 + K0 + 4);
            float4 f2 = *(const float4*)(qrow + d0 + 16 + K0);
            float4 f3 = *(const float4*)(qrow + d0 + 16 + K0 + 4);
            union { v16bf v; v2bf p[8]; v8bf h[2]; } fa;
            fa.p[0] = pk2bf(f0.x * scale, f0.y * scale);
            fa.p[1] = pk2bf(f0.z * scale, f0.w * scale);
            fa.p[2] = pk2bf(f1.x * scale, f1.y * scale);
            fa.p[3] = pk2bf(f1.z * scale, f1.w * scale);
            fa.p[4] = pk2bf(f2.x * scale, f2.y * scale);
            fa.p[5] = pk2bf(f2.z * scale, f2.w * scale);
            fa.p[6] = pk2bf(f3.x * scale, f3.y * scale);
            fa.p[7] = pk2bf(f3.z * scale, f3.w * scale);
            v8bf* slot = (v8bf*)&qlds[((wave * 8 + t * 4 + c) * 32 + lane) * 16];
            slot[0] = fa.h[0];
            slot[1] = fa.h[1];
        }
    }

    v8f acc[16];                 // O tile: 32 rows x 128 cols; [t*8+dt]
    v8f zero = {0.f,0.f,0.f,0.f,0.f,0.f,0.f,0.f};
#pragma unroll
    for (int i = 0; i < 16; ++i) acc[i] = zero;
    float lsum[16];
#pragma unroll
    for (int i = 0; i < 16; ++i) lsum[i] = 0.f;

    const float* kbase = kp + (size_t)b * LK * DD;
    const float* vbase = vp + (size_t)b * LK * DD;

    // Mask: whole tensor is 2^27 bytes, so offsets fit in u32. Keep maskp's
    // global provenance (global_load_u8, not flat) and make the wave-uniform
    // row base a readfirstlane'd scalar; per-lane part is one 32-bit voffset,
    // per-(t,r) row stride folds into the 24-bit instruction offset.
    const unsigned mrow0 =
        __builtin_amdgcn_readfirstlane((unsigned)(b * LQ + qbase) * (unsigned)LK);
    const int mvoff = hi * 8 * LK + l16;

    // cooperative-load indexing: 256 threads cover a 32x128 fp32 tile
    const int crow = tid >> 3;          // 0..31  (key row in tile)
    const int ccol = (tid & 7) * 16;    // 0,16,...,112

    for (int kb = 0; kb < LK; kb += 32) {
        __syncthreads();                  // previous compute done reading LDS
        // ---- stage K (row-major) and V (transposed) as bf16 in LDS ----
        {
            const float* ks = kbase + (size_t)(kb + crow) * DD + ccol;
            const float* vs = vbase + (size_t)(kb + crow) * DD + ccol;
#pragma unroll
            for (int i = 0; i < 4; ++i) {
                float4 f = *(const float4*)(ks + i * 4);
                union { v4bf v; v2bf p[2]; } bv;
                bv.p[0] = pk2bf(f.x, f.y);
                bv.p[1] = pk2bf(f.z, f.w);
                *(v4bf*)&kbuf[crow * 128 + ccol + i * 4] = bv.v;

                float4 g = *(const float4*)(vs + i * 4);
                v2bf g01 = pk2bf(g.x, g.y);
                v2bf g23 = pk2bf(g.z, g.w);
                const int d = ccol + i * 4;
                vbuf[(d + 0) * 32 + crow] = g01[0];
                vbuf[(d + 1) * 32 + crow] = g01[1];   // ds_store_b16_d16_hi
                vbuf[(d + 2) * 32 + crow] = g23[0];
                vbuf[(d + 3) * 32 + crow] = g23[1];
            }
        }
        __syncthreads();

        // ---- gemm1: S = Q * K^T, 32 q rows x 32 keys ----
        // s0: rows 0-15 keys kb..+16   s1: rows 0-15 keys kb+16..+32
        // s2: rows 16-31 keys kb..+16  s3: rows 16-31 keys kb+16..+32
        v8f s0 = zero, s1 = zero, s2 = zero, s3 = zero;
#pragma unroll
        for (int c = 0; c < 4; ++c) {
            const int d0 = c * 32;
            union { v16bf v; v8bf h[2]; } qa0, qa1, b0, b1;
            const v8bf* q0 = (const v8bf*)&qlds[((wave * 8 + c) * 32 + lane) * 16];
            const v8bf* q1 = (const v8bf*)&qlds[((wave * 8 + 4 + c) * 32 + lane) * 16];
            qa0.h[0] = q0[0];  qa0.h[1] = q0[1];
            qa1.h[0] = q1[0];  qa1.h[1] = q1[1];
            const __bf16* k0 = &kbuf[l16 * 128 + d0 + hi * 8];
            const __bf16* k1 = &kbuf[(16 + l16) * 128 + d0 + hi * 8];
            b0.h[0] = *(const v8bf*)(k0);     b0.h[1] = *(const v8bf*)(k0 + 16);
            b1.h[0] = *(const v8bf*)(k1);     b1.h[1] = *(const v8bf*)(k1 + 16);
            s0 = __builtin_amdgcn_wmma_f32_16x16x32_bf16(false, qa0.v, false, b0.v,
                                                         (short)0, s0, false, false);
            s1 = __builtin_amdgcn_wmma_f32_16x16x32_bf16(false, qa0.v, false, b1.v,
                                                         (short)0, s1, false, false);
            s2 = __builtin_amdgcn_wmma_f32_16x16x32_bf16(false, qa1.v, false, b0.v,
                                                         (short)0, s2, false, false);
            s3 = __builtin_amdgcn_wmma_f32_16x16x32_bf16(false, qa1.v, false, b1.v,
                                                         (short)0, s3, false, false);
        }

        // ---- exp2, mask, row-sum partials, stage P as bf16 ----
#pragma unroll
        for (int t = 0; t < 2; ++t) {
            const v8f& sa = t ? s2 : s0;
            const v8f& sb = t ? s3 : s1;
#pragma unroll
            for (int r = 0; r < 8; ++r) {
                const int m = r + hi * 8;                   // row within 16-tile
                const unsigned mo = mrow0 + (unsigned)(kb + (t * 16 + r) * LK);
                const int m0 = maskp[mo + mvoff];           // global_load_u8
                const int m1 = maskp[mo + mvoff + 16];
                float p0 = fast_exp2(sa[r]);                // log2e folded into Q
                float p1 = fast_exp2(sb[r]);
                p0 = m0 ? 0.0f : p0;
                p1 = m1 ? 0.0f : p1;
                lsum[t * 8 + r] += p0 + p1;
                v2bf pk = pk2bf(p0, p1);
                __bf16* pr = &pbuf[wave][(t * 16 + m) * 32];
                pr[l16]      = pk[0];
                pr[16 + l16] = pk[1];                       // ds_store_b16_d16_hi
            }
        }
        asm volatile("s_wait_dscnt 0" ::: "memory");        // P stores -> A loads

        // ---- P tiles (2 x 16x32) = two A fragments ----
        union { v16bf v; v8bf h[2]; } pa0, pa1;
        {
            const __bf16* pp0 = &pbuf[wave][l16 * 32 + hi * 8];
            const __bf16* pp1 = &pbuf[wave][(16 + l16) * 32 + hi * 8];
            pa0.h[0] = *(const v8bf*)(pp0);  pa0.h[1] = *(const v8bf*)(pp0 + 16);
            pa1.h[0] = *(const v8bf*)(pp1);  pa1.h[1] = *(const v8bf*)(pp1 + 16);
        }

        // ---- gemm2: O += P * V; each V fragment feeds both row-halves ----
#pragma unroll
        for (int dt = 0; dt < 8; ++dt) {
            union { v16bf v; v8bf h[2]; } vb;
            const __bf16* vv = &vbuf[(dt * 16 + l16) * 32 + hi * 8];
            vb.h[0] = *(const v8bf*)(vv);
            vb.h[1] = *(const v8bf*)(vv + 16);
            acc[dt]     = __builtin_amdgcn_wmma_f32_16x16x32_bf16(false, pa0.v, false, vb.v,
                                                                  (short)0, acc[dt],     false, false);
            acc[8 + dt] = __builtin_amdgcn_wmma_f32_16x16x32_bf16(false, pa1.v, false, vb.v,
                                                                  (short)0, acc[8 + dt], false, false);
        }
    }

    // ---- reduce row sums within each 16-lane half, normalize, store ----
    float inv[16];
#pragma unroll
    for (int i = 0; i < 16; ++i) {
        float s = lsum[i];
        s += __shfl_xor(s, 1, 32);
        s += __shfl_xor(s, 2, 32);
        s += __shfl_xor(s, 4, 32);
        s += __shfl_xor(s, 8, 32);
        inv[i] = 1.0f / s;
    }
    float* ob = outp + ((size_t)b * LQ + qbase) * DD;
#pragma unroll
    for (int t = 0; t < 2; ++t) {
#pragma unroll
        for (int dt = 0; dt < 8; ++dt) {
#pragma unroll
            for (int r = 0; r < 8; ++r) {
                const int m = r + hi * 8;
                ob[(size_t)(t * 16 + m) * DD + dt * 16 + l16] =
                    acc[t * 8 + dt][r] * inv[t * 8 + r];
            }
        }
    }
}

extern "C" void kernel_launch(void* const* d_in, const int* in_sizes, int n_in,
                              void* d_out, int out_size, void* d_ws, size_t ws_size,
                              hipStream_t stream) {
    const float* q = (const float*)d_in[0];
    const float* k = (const float*)d_in[1];
    const float* v = (const float*)d_in[2];
    const unsigned char* mask = (const unsigned char*)d_in[3];
    float* out = (float*)d_out;

    dim3 grid(NB * (LQ / 256));   // 256 blocks: (batch, 256-row q block)
    dim3 block(256);              // 8 waves; one 32-row q tile per wave
    attn_wmma_kernel<<<grid, block, 0, stream>>>(q, k, v, mask, out);
}
